// MultiHeadAttention_66881230733593
// MI455X (gfx1250) — matrix operand order repaired and checked by reading, not compiled, not run on hardware
//
#include <hip/hip_runtime.h>
#include <hip/hip_bf16.h>
#include <math.h>

#define D_MODEL   1024
#define NUM_HEADS 16
#define HEAD_DIM  64
#define BATCH     2
#define SEQ       2048
#define MTOT      (BATCH*SEQ)   // 4096 rows for the projection GEMMs

#define KB        64            // keys per attention block
#define LSTRIDE   66            // 64 + 1 dword pad -> 33-bank row stride (conflict-free)

// 1/sqrt(HEAD_DIM) * log2(e): folded into the Q projection so the attention
// inner loop uses native exp2 with no extra scaling multiplies.
#define Q_SCALE   0.18033688f

typedef __attribute__((ext_vector_type(16))) __bf16 v16bf;
typedef __attribute__((ext_vector_type(8)))  __bf16 bf8v;
typedef __attribute__((ext_vector_type(2)))  __bf16 bf2;
typedef __attribute__((ext_vector_type(8)))  float  v8f;

#if defined(__has_builtin)
#  if __has_builtin(__builtin_amdgcn_tensor_load_to_lds)
#    define HAVE_TDM 1
#  endif
#endif
#ifndef HAVE_TDM
#  define HAVE_TDM 0
#endif

__device__ __forceinline__ __bf16 f2bf(float f) {
  unsigned u = __builtin_bit_cast(unsigned, f);
  u += 0x7FFFu + ((u >> 16) & 1u);          // round-to-nearest-even
  unsigned short h = (unsigned short)(u >> 16);
  return __builtin_bit_cast(__bf16, h);
}

__device__ __forceinline__ v8f wmma_bf16(v16bf a, v16bf b, v8f c) {
  return __builtin_amdgcn_wmma_f32_16x16x32_bf16(false, a, false, b, (short)0, c,
                                                 false, false);
}

// ---- 16-lane butterfly reductions in pure VALU via DPP16 ------------------
// xor1/xor2 via quad_perm, xor4 via ROW_HALF_MIRROR, xor8 via ROW_MIRROR.
// dpp_ctrl must be a frontend constant -> template non-type parameter.
template <int CTRL>
__device__ __forceinline__ float dpp_mov(float x) {
  int i = __builtin_bit_cast(int, x);
  int r = __builtin_amdgcn_update_dpp(i, i, CTRL, 0xF, 0xF, true);
  return __builtin_bit_cast(float, r);
}
__device__ __forceinline__ float red16_max(float x) {
  x = fmaxf(x, dpp_mov<0xB1>(x));    // quad_perm [1,0,3,2]  (xor 1)
  x = fmaxf(x, dpp_mov<0x4E>(x));    // quad_perm [2,3,0,1]  (xor 2)
  x = fmaxf(x, dpp_mov<0x141>(x));   // ROW_HALF_MIRROR      (xor 4-level)
  x = fmaxf(x, dpp_mov<0x140>(x));   // ROW_MIRROR           (xor 8-level)
  return x;
}
__device__ __forceinline__ float red16_sum(float x) {
  x += dpp_mov<0xB1>(x);
  x += dpp_mov<0x4E>(x);
  x += dpp_mov<0x141>(x);
  x += dpp_mov<0x140>(x);
  return x;
}

__device__ __forceinline__ float fast_exp2(float x) {
#if defined(__has_builtin)
#  if __has_builtin(__builtin_amdgcn_exp2f)
  return __builtin_amdgcn_exp2f(x);
#  else
  return exp2f(x);
#  endif
#else
  return exp2f(x);
#endif
}

#if HAVE_TDM
typedef __attribute__((ext_vector_type(4))) unsigned v4u;
typedef __attribute__((ext_vector_type(8))) int      v8i;
typedef __attribute__((ext_vector_type(4))) int      v4i;

// 2D TDM load: tile rows x cols (bf16) global -> LDS, 1 dword LDS padding
// after every 32 dwords (one 64-elem row) => LDS row stride LSTRIDE elems.
__device__ __forceinline__ void tdm_load_2d_padded(unsigned lds_off, const void* gptr,
                                                   unsigned rows, unsigned cols,
                                                   unsigned row_stride_elems)
{
  unsigned long long ga = (unsigned long long)gptr;
  v4u g0;
  g0[0] = 1u;                                             // count=1, user mode
  g0[1] = lds_off;                                        // lds_addr (bytes)
  g0[2] = (unsigned)(ga & 0xFFFFFFFFull);                 // global_addr[31:0]
  g0[3] = (unsigned)((ga >> 32) & 0x1FFFFFFull)           // global_addr[56:32]
        | 0x80000000u;                                    // type=2 ("image")
  v8i g1;
  g1[0] = (int)((1u << 16)      // data_size = 2 bytes
              | (1u << 20)      // pad_enable
              | (4u << 22)      // pad_interval: every 32 dwords (one row)
              | (0u << 25));    // pad_amount: 1 dword
  g1[1] = (int)(cols << 16);             // tensor_dim0[15:0]
  g1[2] = (int)(rows << 16);             // tensor_dim0[31:16]=0 | tensor_dim1[15:0]
  g1[3] = (int)(cols << 16);             // tensor_dim1[31:16]=0 | tile_dim0
  g1[4] = (int)rows;                     // tile_dim1 | tile_dim2=0
  g1[5] = (int)row_stride_elems;         // tensor_dim0_stride[31:0]
  g1[6] = 0;
  g1[7] = 0;
  v4i z4 = {0, 0, 0, 0};
#if __clang_major__ >= 23
  v8i z8 = {0, 0, 0, 0, 0, 0, 0, 0};
  __builtin_amdgcn_tensor_load_to_lds(g0, g1, z4, z4, z8, 0);
#else
  __builtin_amdgcn_tensor_load_to_lds(g0, g1, z4, z4, 0);
#endif
}
#endif  // HAVE_TDM

// ---------------------------------------------------------------------------
// Elementwise f32 -> bf16 conversion (one-time; keeps cvt out of GEMM loop)
// ---------------------------------------------------------------------------
__global__ void __launch_bounds__(256)
cvt_f32_bf16_kernel(const float* __restrict__ in, __bf16* __restrict__ out, int n)
{
  const int i = (blockIdx.x * 256 + threadIdx.x) * 4;
  if (i < n) {
    float4 v = *(const float4*)(in + i);
    out[i + 0] = f2bf(v.x);
    out[i + 1] = f2bf(v.y);
    out[i + 2] = f2bf(v.z);
    out[i + 3] = f2bf(v.w);
  }
}

// ---------------------------------------------------------------------------
// GEMM: C[M,N] = (A[M,K] @ W[N,K]^T + bias) * cscale, bf16 operands, f32 acc.
// Block 128 = 4 waves (2x2); wave tile 32(M) x 64(N): 8 WMMAs per K-step.
// ---------------------------------------------------------------------------
template <bool HEADSPLIT>
__global__ void __launch_bounds__(128)
gemm_bf16_wmma_kernel(const __bf16* __restrict__ A, const __bf16* __restrict__ W,
                      const float* __restrict__ bias, void* __restrict__ out_,
                      float cscale)
{
  const int lane = threadIdx.x & 31;
  const int wave = threadIdx.x >> 5;
  const int r    = lane & 15;
  const int half = lane >> 4;
  const int wm   = wave >> 1, wn = wave & 1;
  const int mBase = blockIdx.y * 64 + wm * 32;
  const int nBase = blockIdx.x * 128 + wn * 64;

  v8f c[2][4] = {};

  for (int kk = 0; kk < D_MODEL; kk += 32) {
    v16bf a[2];
    #pragma unroll
    for (int mt = 0; mt < 2; ++mt) {
      const int row = mBase + mt * 16 + r;
      const __bf16* ar = A + (size_t)row * D_MODEL + kk;
      #pragma unroll
      for (int i = 0; i < 8; ++i) {
        const int k = 2 * i + 16 * half;
        bf2 p = *(const bf2*)(ar + k);
        a[mt][2 * i] = p[0]; a[mt][2 * i + 1] = p[1];
      }
    }
    v16bf b[4];
    #pragma unroll
    for (int nt = 0; nt < 4; ++nt) {
      const int col = nBase + nt * 16 + r;
      const __bf16* wr = W + (size_t)col * D_MODEL + kk;
      #pragma unroll
      for (int i = 0; i < 8; ++i) {
        const int k = 2 * i + 16 * half;
        bf2 p = *(const bf2*)(wr + k);
        b[nt][2 * i] = p[0]; b[nt][2 * i + 1] = p[1];
      }
    }
    #pragma unroll
    for (int mt = 0; mt < 2; ++mt)
      #pragma unroll
      for (int nt = 0; nt < 4; ++nt)
        c[mt][nt] = wmma_bf16(a[mt], b[nt], c[mt][nt]);
  }

  #pragma unroll
  for (int mt = 0; mt < 2; ++mt) {
    #pragma unroll
    for (int nt = 0; nt < 4; ++nt) {
      const int n0 = nBase + nt * 16;
      #pragma unroll
      for (int j = 0; j < 8; ++j) {
        const int row = mBase + mt * 16 + j + 8 * half;
        const int col = n0 + r;
        const float v = (c[mt][nt][j] + bias[col]) * cscale;
        if (HEADSPLIT) {
          __bf16* outb = (__bf16*)out_;
          const int bb = row >> 11, ss = row & (SEQ - 1);
          const int hh = col >> 6,  dd = col & (HEAD_DIM - 1);
          outb[(((size_t)bb * NUM_HEADS + hh) * SEQ + ss) * HEAD_DIM + dd] = f2bf(v);
        } else {
          ((float*)out_)[(size_t)row * D_MODEL + col] = v;
        }
      }
    }
  }
}

// ---------------------------------------------------------------------------
// Flash attention, 64-key blocks. Per (b,h): 4 waves x 16 query rows.
// K staged via TDM (padded rows), V staged transposed by the workgroup.
// Scores arrive pre-scaled by log2(e)/sqrt(Hd); softmax runs in base 2 with
// DPP16 butterfly reductions (no LDS traffic for reductions).
// ---------------------------------------------------------------------------
__global__ void __launch_bounds__(128)
flash_attn_kernel(const __bf16* __restrict__ Q, const __bf16* __restrict__ K,
                  const __bf16* __restrict__ V, __bf16* __restrict__ AO)
{
  __shared__ __bf16 Klds[KB * LSTRIDE];      // [key][d], padded rows
  __shared__ __bf16 Vt[HEAD_DIM * LSTRIDE];  // [d][key], padded rows
  __shared__ __bf16 Plds[4][16 * LSTRIDE];   // per-wave P staging, padded rows

  const int lane = threadIdx.x & 31;
  const int wave = threadIdx.x >> 5;
  const int r    = lane & 15;
  const int half = lane >> 4;
  const int b    = blockIdx.z, h = blockIdx.y;
  const int m0   = blockIdx.x * 64 + wave * 16;

  const size_t headOff = ((size_t)b * NUM_HEADS + h) * SEQ * HEAD_DIM;
  const __bf16* Qh = Q + headOff;
  const __bf16* Kh = K + headOff;
  const __bf16* Vh = V + headOff;
  __bf16* Pw = &Plds[wave][0];

#if HAVE_TDM
  const unsigned kldsOff = (unsigned)(unsigned long long)(uintptr_t)&Klds[0];
#endif

  // Q A-fragments for d in [0,32) and [32,64); Q already carries Q_SCALE
  v16bf qa0, qa1;
  {
    const __bf16* qrow = Qh + (size_t)(m0 + r) * HEAD_DIM;
    #pragma unroll
    for (int i = 0; i < 8; ++i) {
      const int k = 2 * i + 16 * half;
      bf2 p0 = *(const bf2*)(qrow + k);
      bf2 p1 = *(const bf2*)(qrow + 32 + k);
      qa0[2 * i] = p0[0]; qa0[2 * i + 1] = p0[1];
      qa1[2 * i] = p1[0]; qa1[2 * i + 1] = p1[1];
    }
  }

  v8f O[4] = {};
  float mrow[8], lrow[8];
  #pragma unroll
  for (int j = 0; j < 8; ++j) { mrow[j] = -INFINITY; lrow[j] = 0.f; }

  for (int blkk = 0; blkk < SEQ / KB; ++blkk) {
    const int k0 = blkk * KB;

    __syncthreads();   // previous block fully consumed

#if HAVE_TDM
    if (wave == 0)     // one DMA per workgroup; EXEC ignored by TDM
      tdm_load_2d_padded(kldsOff, Kh + (size_t)k0 * HEAD_DIM, KB, HEAD_DIM, HEAD_DIM);
#else
    {   // cooperative K copy: thread -> one 32-elem row segment
      const int t = threadIdx.x, keyl = t >> 1, e0 = (t & 1) * 32;
      const __bf16* src = Kh + (size_t)(k0 + keyl) * HEAD_DIM + e0;
      #pragma unroll
      for (int q8 = 0; q8 < 4; ++q8) {
        bf8v v0 = *(const bf8v*)(src + q8 * 8);
        #pragma unroll
        for (int e = 0; e < 8; e += 2) {
          bf2 p; p[0] = v0[e]; p[1] = v0[e + 1];
          *(bf2*)&Klds[keyl * LSTRIDE + e0 + q8 * 8 + e] = p;
        }
      }
    }
#endif
    {   // cooperative transposed V copy: Vt[d][key] = V[key][d]
      const int t = threadIdx.x, keyl = t >> 1, d0 = (t & 1) * 32;
      const __bf16* src = Vh + (size_t)(k0 + keyl) * HEAD_DIM + d0;
      if (blkk + 1 < SEQ / KB)
        __builtin_prefetch(src + (size_t)KB * HEAD_DIM, 0, 1);
      #pragma unroll
      for (int q8 = 0; q8 < 4; ++q8) {
        bf8v v0 = *(const bf8v*)(src + q8 * 8);
        #pragma unroll
        for (int e = 0; e < 8; ++e)
          Vt[(d0 + q8 * 8 + e) * LSTRIDE + keyl] = v0[e];
      }
    }
#if HAVE_TDM
    if (wave == 0) __builtin_amdgcn_s_wait_tensorcnt(0);
#endif
    __syncthreads();   // staged K/V visible to all waves

    // Scores: four 16(q) x 16(key) tiles; B[k=d][n=key] = Klds[key][d]
    v8f st[4];
    #pragma unroll
    for (int nt = 0; nt < 4; ++nt) {
      v16bf kf0, kf1;
      const __bf16* krow = &Klds[(nt * 16 + r) * LSTRIDE];
      #pragma unroll
      for (int i = 0; i < 8; ++i) {
        const int k = 2 * i + 16 * half;
        bf2 p0 = *(const bf2*)(krow + k);
        bf2 p1 = *(const bf2*)(krow + 32 + k);
        kf0[2 * i] = p0[0]; kf0[2 * i + 1] = p0[1];
        kf1[2 * i] = p1[0]; kf1[2 * i + 1] = p1[1];
      }
      v8f s = {};
      s = wmma_bf16(qa0, kf0, s);
      s = wmma_bf16(qa1, kf1, s);
      st[nt] = s;
    }

    // Online softmax (base-2); DPP16 butterflies, rows = half-wave groups
    #pragma unroll
    for (int j = 0; j < 8; ++j) {
      const float s0 = st[0][j], s1 = st[1][j], s2 = st[2][j], s3 = st[3][j];
      float mx = fmaxf(fmaxf(s0, s1), fmaxf(s2, s3));
      mx = red16_max(mx);
      const float nm = fmaxf(mrow[j], mx);
      const float f  = fast_exp2(mrow[j] - nm);
      const float p0 = fast_exp2(s0 - nm);
      const float p1 = fast_exp2(s1 - nm);
      const float p2 = fast_exp2(s2 - nm);
      const float p3 = fast_exp2(s3 - nm);
      float rs = (p0 + p1) + (p2 + p3);
      rs = red16_sum(rs);
      lrow[j] = lrow[j] * f + rs;
      mrow[j] = nm;
      #pragma unroll
      for (int t = 0; t < 4; ++t) O[t][j] *= f;
      const int prow = j + 8 * half;
      Pw[prow * LSTRIDE + r]      = f2bf(p0);
      Pw[prow * LSTRIDE + 16 + r] = f2bf(p1);
      Pw[prow * LSTRIDE + 32 + r] = f2bf(p2);
      Pw[prow * LSTRIDE + 48 + r] = f2bf(p3);
    }

    // P as two 16x32 A-fragments (row remap via per-wave LDS)
    v16bf pa0, pa1;
    {
      const __bf16* pr = Pw + r * LSTRIDE;
      #pragma unroll
      for (int i = 0; i < 8; ++i) {
        const int k = 2 * i + 16 * half;
        bf2 p0 = *(const bf2*)(pr + k);
        bf2 p1 = *(const bf2*)(pr + 32 + k);
        pa0[2 * i] = p0[0]; pa0[2 * i + 1] = p0[1];
        pa1[2 * i] = p1[0]; pa1[2 * i + 1] = p1[1];
      }
    }

    // O(16x64) += P(16x64) @ V(64x64); B[k=key][n=d] = Vt[d][key]
    #pragma unroll
    for (int t = 0; t < 4; ++t) {
      v16bf vb0, vb1;
      const __bf16* vr = &Vt[(t * 16 + r) * LSTRIDE];
      #pragma unroll
      for (int i = 0; i < 8; ++i) {
        const int k = 2 * i + 16 * half;
        bf2 p0 = *(const bf2*)(vr + k);        // keys 0..31
        bf2 p1 = *(const bf2*)(vr + 32 + k);   // keys 32..63
        vb0[2 * i] = p0[0]; vb0[2 * i + 1] = p0[1];
        vb1[2 * i] = p1[0]; vb1[2 * i + 1] = p1[1];
      }
      O[t] = wmma_bf16(pa0, vb0, O[t]);
      O[t] = wmma_bf16(pa1, vb1, O[t]);
    }
  }

  // Normalize, store concat layout [B, S, H*Hd] (bf16)
  #pragma unroll
  for (int j = 0; j < 8; ++j) {
    const float inv = 1.0f / lrow[j];
    const int qi = m0 + j + 8 * half;
    const size_t base = ((size_t)b * SEQ + qi) * D_MODEL + (size_t)h * HEAD_DIM;
    #pragma unroll
    for (int t = 0; t < 4; ++t)
      AO[base + t * 16 + r] = f2bf(O[t][j] * inv);
  }
}

// ---------------------------------------------------------------------------
extern "C" void kernel_launch(void* const* d_in, const int* in_sizes, int n_in,
                              void* d_out, int out_size, void* d_ws, size_t ws_size,
                              hipStream_t stream) {
  const float* query = (const float*)d_in[0];
  const float* key_  = (const float*)d_in[1];
  const float* value = (const float*)d_in[2];
  const float* W_q   = (const float*)d_in[3];
  const float* b_q   = (const float*)d_in[4];
  const float* W_k   = (const float*)d_in[5];
  const float* b_k   = (const float*)d_in[6];
  const float* W_v   = (const float*)d_in[7];
  const float* b_v   = (const float*)d_in[8];
  const float* W_o   = (const float*)d_in[9];
  const float* b_o   = (const float*)d_in[10];

  const size_t xElems = (size_t)MTOT * D_MODEL;       // 4M
  const size_t wElems = (size_t)D_MODEL * D_MODEL;    // 1M
  char* p = (char*)d_ws;
  __bf16* Xq  = (__bf16*)p; p += xElems * 2;
  __bf16* Xk  = (__bf16*)p; p += xElems * 2;
  __bf16* Xv  = (__bf16*)p; p += xElems * 2;
  __bf16* Wqb = (__bf16*)p; p += wElems * 2;
  __bf16* Wkb = (__bf16*)p; p += wElems * 2;
  __bf16* Wvb = (__bf16*)p; p += wElems * 2;
  __bf16* Wob = (__bf16*)p; p += wElems * 2;
  __bf16* Qw  = (__bf16*)p; p += xElems * 2;
  __bf16* Kw  = (__bf16*)p; p += xElems * 2;
  __bf16* Vw  = (__bf16*)p; p += xElems * 2;
  __bf16* AO  = (__bf16*)p; p += xElems * 2;

  dim3 cblk(256);
  cvt_f32_bf16_kernel<<<dim3(xElems / 4 / 256), cblk, 0, stream>>>(query, Xq, (int)xElems);
  cvt_f32_bf16_kernel<<<dim3(xElems / 4 / 256), cblk, 0, stream>>>(key_,  Xk, (int)xElems);
  cvt_f32_bf16_kernel<<<dim3(xElems / 4 / 256), cblk, 0, stream>>>(value, Xv, (int)xElems);
  cvt_f32_bf16_kernel<<<dim3(wElems / 4 / 256), cblk, 0, stream>>>(W_q, Wqb, (int)wElems);
  cvt_f32_bf16_kernel<<<dim3(wElems / 4 / 256), cblk, 0, stream>>>(W_k, Wkb, (int)wElems);
  cvt_f32_bf16_kernel<<<dim3(wElems / 4 / 256), cblk, 0, stream>>>(W_v, Wvb, (int)wElems);
  cvt_f32_bf16_kernel<<<dim3(wElems / 4 / 256), cblk, 0, stream>>>(W_o, Wob, (int)wElems);

  dim3 blk(128);
  dim3 gGemm(D_MODEL / 128, MTOT / 64);       // (8, 64)
  gemm_bf16_wmma_kernel<true><<<gGemm, blk, 0, stream>>>(Xq, Wqb, b_q, Qw, Q_SCALE);
  gemm_bf16_wmma_kernel<true><<<gGemm, blk, 0, stream>>>(Xk, Wkb, b_k, Kw, 1.0f);
  gemm_bf16_wmma_kernel<true><<<gGemm, blk, 0, stream>>>(Xv, Wvb, b_v, Vw, 1.0f);

  dim3 gAttn(SEQ / 64, NUM_HEADS, BATCH);     // (32, 16, 2)
  flash_attn_kernel<<<gAttn, blk, 0, stream>>>(Qw, Kw, Vw, AO);

  gemm_bf16_wmma_kernel<false><<<gGemm, blk, 0, stream>>>(AO, Wob, b_o, (float*)d_out, 1.0f);
}
